// CascadedGroupAttention_11871289606419
// MI455X (gfx1250) — matrix operand order repaired and checked by reading, not compiled, not run on hardware
//
#include <hip/hip_runtime.h>
#include <hip/hip_bf16.h>

// CDNA5 / gfx1250: wave32, WMMA f32 = f16(16x32) x f16(32x16) + f32
typedef _Float16 v8h  __attribute__((ext_vector_type(8)));
typedef _Float16 v16h __attribute__((ext_vector_type(16)));
typedef float    v8f  __attribute__((ext_vector_type(8)));

#define CGA_B     512
#define CGA_DIM   256
#define CGA_HEADS 4
#define CGA_KD    16
#define CGA_VD    64
#define CGA_QKV   96      // 2*KD + VD
#define CGA_N     196     // 14*14
#define CGA_NP    208     // padded rows/cols to 13*16
#define CGA_NK    224     // AV K-dim padded to 7*32
#define CGA_RES   14

__device__ __forceinline__ v8f wmma_f16x32(v16h a, v16h b, v8f c) {
  // (neg_a, A, neg_b, B, c_mod, C, reuse_a, reuse_b)
  return __builtin_amdgcn_wmma_f32_16x16x32_f16(false, a, false, b, (short)0, c,
                                                false, false);
}

// A operand, 16-bit, 16x32 tile: lane L (lo=L&15 is M-row, hi=L>>4):
//   a[i] = A[lo][k0 + 16*hi + i], i=0..15
__device__ __forceinline__ v16h load_a16(const _Float16* row, int k0, int hi) {
  return *(const v16h*)(row + k0 + 16 * hi);
}
// B operand, 16-bit, 32x16 tile: b[i] = B[k0 + 16*hi + i][lo], i=0..15.
// With B stored transposed as BT[n][k], that is 16 contiguous halfs.
__device__ __forceinline__ v16h load_bt16(const _Float16* rowT, int k0, int hi) {
  return *(const v16h*)(rowT + k0 + 16 * hi);
}

// ---------------------------------------------------------------------------
// Kernel 1: full cascaded attention for one batch element per workgroup.
// Writes relu(concat(head outputs)) into out_stage (=d_out) as (B,256,196).
// ---------------------------------------------------------------------------
__global__ __launch_bounds__(256)
void cga_attn_kernel(const float* __restrict__ x,
                     const float* __restrict__ qkv_w,
                     const float* __restrict__ qkv_g,
                     const float* __restrict__ qkv_b,
                     const float* __restrict__ qkv_rm,
                     const float* __restrict__ qkv_rv,
                     const float* __restrict__ dw_w,
                     const float* __restrict__ dw_g,
                     const float* __restrict__ dw_b,
                     const float* __restrict__ dw_rm,
                     const float* __restrict__ dw_rv,
                     const float* __restrict__ attn_biases,
                     const int*   __restrict__ bias_idxs,
                     float* __restrict__ out_stage) {
  // ~205 KB LDS total
  __shared__ __attribute__((aligned(32))) _Float16 sFeatT[CGA_NP][CGA_VD]; // feat^T (B of QKV)
  __shared__ __attribute__((aligned(32))) _Float16 sWh[CGA_QKV][CGA_VD];   // qkv_w f16 (A of QKV)
  __shared__ __attribute__((aligned(32))) float    sQ[CGA_KD][CGA_NP];     // q f32 pre-conv
  __shared__ __attribute__((aligned(32))) _Float16 sKT[CGA_NP][32];        // k^T, K-pad 16->32 (B of QK)
  __shared__ __attribute__((aligned(32))) _Float16 sQdT[CGA_NP][32];       // dwconv q ^T (A of QK)
  __shared__ __attribute__((aligned(32))) _Float16 sV[CGA_VD][CGA_NK];     // v (A of AV)
  __shared__ __attribute__((aligned(32))) _Float16 sAttnH[CGA_NP][CGA_NK]; // attn (B of AV)
  __shared__ float sQKVs[CGA_QKV], sQKVsh[CGA_QKV];
  __shared__ float sDWw[CGA_KD * 25];
  __shared__ float sDWs[CGA_KD], sDWsh[CGA_KD];
  __shared__ float sBias[CGA_N];

  const int b    = blockIdx.x;
  const int tid  = threadIdx.x;
  const int lane = tid & 31;
  const int wave = tid >> 5;
  const int lo   = lane & 15;
  const int hi   = lane >> 4;

  const float* xb = x + (size_t)b * CGA_DIM * CGA_N;

  // ---- one-time init: feat^T = chunk0^T, zero all K/N pads ----
  for (int i = tid; i < CGA_VD * CGA_N; i += 256) {
    int c = i / CGA_N, n = i % CGA_N;            // coalesced global read
    sFeatT[n][c] = (_Float16)xb[c * CGA_N + n];
  }
  for (int i = tid; i < (CGA_NP - CGA_N) * CGA_VD; i += 256)
    sFeatT[CGA_N + i / CGA_VD][i % CGA_VD] = (_Float16)0.f;
  for (int i = tid; i < CGA_VD * (CGA_NK - CGA_NP); i += 256)
    sV[i / 16][CGA_NP + (i % 16)] = (_Float16)0.f;
  for (int i = tid; i < CGA_NP * 16; i += 256) {
    sKT[i / 16][16 + (i % 16)]  = (_Float16)0.f;
    sQdT[i / 16][16 + (i % 16)] = (_Float16)0.f;
  }
  for (int i = tid; i < (CGA_NP - CGA_N) * 16; i += 256)
    sQdT[CGA_N + i / 16][i % 16] = (_Float16)0.f;

  for (int h = 0; h < CGA_HEADS; ++h) {
    if (h > 0) {
      // cascade: feat += chunk[h]
      for (int i = tid; i < CGA_VD * CGA_N; i += 256) {
        int c = i / CGA_N, n = i % CGA_N;
        sFeatT[n][c] = (_Float16)((float)sFeatT[n][c] +
                                  xb[(h * CGA_VD + c) * CGA_N + n]);
      }
    }
    // Stage weights (f16) and folded BN params.
    for (int i = tid; i < CGA_QKV * CGA_VD; i += 256)
      sWh[i / CGA_VD][i % CGA_VD] = (_Float16)qkv_w[h * CGA_QKV * CGA_VD + i];
    for (int i = tid; i < CGA_QKV; i += 256) {
      float s = qkv_g[h * CGA_QKV + i] * rsqrtf(qkv_rv[h * CGA_QKV + i] + 1e-5f);
      sQKVs[i]  = s;
      sQKVsh[i] = qkv_b[h * CGA_QKV + i] - qkv_rm[h * CGA_QKV + i] * s;
    }
    for (int i = tid; i < CGA_KD * 25; i += 256)
      sDWw[i] = dw_w[h * CGA_KD * 25 + i];
    for (int i = tid; i < CGA_KD; i += 256) {
      float s = dw_g[h * CGA_KD + i] * rsqrtf(dw_rv[h * CGA_KD + i] + 1e-5f);
      sDWs[i]  = s;
      sDWsh[i] = dw_b[h * CGA_KD + i] - dw_rm[h * CGA_KD + i] * s;
    }
    for (int i = tid; i < CGA_N; i += 256)
      sBias[i] = attn_biases[h * CGA_N + i];
    // Prefetch next head's cascade chunk while we compute this one.
    if (h + 1 < CGA_HEADS) {
      const float* nxt = xb + (size_t)(h + 1) * CGA_VD * CGA_N;
      for (int i = tid * 32; i < CGA_VD * CGA_N; i += 256 * 32)
        __builtin_prefetch(nxt + i, 0, 1);
    }
    __syncthreads();

    // ---- P1: QKV GEMM  f[o,n] = BN(sum_c W[o,c] * feat[c,n]) ----
    // M=96 (6 tiles), N=208 (13 tiles), K=64 (2 wmma steps)
    for (int t = wave; t < 6 * 13; t += 8) {
      int mt = t / 13, nt = t % 13;
      v8f acc = {};
#pragma unroll
      for (int k0 = 0; k0 < CGA_VD; k0 += 32) {
        v16h a  = load_a16(&sWh[mt * 16 + lo][0], k0, hi);
        v16h bb = load_bt16(&sFeatT[nt * 16 + lo][0], k0, hi);
        acc = wmma_f16x32(a, bb, acc);
      }
      int n = nt * 16 + lo;
      if (mt == 0) {            // q rows 0..15 -> f32 buffer for dwconv
#pragma unroll
        for (int r = 0; r < 8; ++r) {
          int m = r + 8 * hi;
          sQ[m][n] = acc[r] * sQKVs[m] + sQKVsh[m];
        }
      } else if (mt == 1) {     // k rows 16..31 -> k^T f16
#pragma unroll
        for (int r = 0; r < 8; ++r) {
          int m = 16 + r + 8 * hi;
          sKT[n][m - 16] = (_Float16)(acc[r] * sQKVs[m] + sQKVsh[m]);
        }
      } else {                  // v rows 32..95 -> v f16
#pragma unroll
        for (int r = 0; r < 8; ++r) {
          int m = mt * 16 + r + 8 * hi;
          sV[m - 32][n] = (_Float16)(acc[r] * sQKVs[m] + sQKVsh[m]);
        }
      }
    }
    __syncthreads();

    // ---- P2: depthwise 5x5 conv on q + BN; fold 1/sqrt(KD)=0.25 ----
    for (int i = tid; i < CGA_KD * CGA_N; i += 256) {
      int c = i / CGA_N, p = i % CGA_N;
      int py = p / CGA_RES, px = p % CGA_RES;
      float acc = 0.f;
#pragma unroll
      for (int ky = 0; ky < 5; ++ky) {
        int yy = py + ky - 2;
        if (yy < 0 || yy >= CGA_RES) continue;
#pragma unroll
        for (int kx = 0; kx < 5; ++kx) {
          int xx = px + kx - 2;
          if (xx < 0 || xx >= CGA_RES) continue;
          acc += sDWw[c * 25 + ky * 5 + kx] * sQ[c][yy * CGA_RES + xx];
        }
      }
      sQdT[p][c] = (_Float16)((acc * sDWs[c] + sDWsh[c]) * 0.25f);
    }
    __syncthreads();

    // ---- P3: attn[n,m] = q^T k + bias   13x13 tiles, K=32 (zero-padded) ----
    for (int t = wave; t < 13 * 13; t += 8) {
      int ntile = t / 13, mtile = t % 13;
      v16h a  = load_a16(&sQdT[ntile * 16 + lo][0], 0, hi);   // A[n][d]
      v16h bb = load_bt16(&sKT[mtile * 16 + lo][0], 0, hi);   // B[d][m] via k^T
      v8f acc = {};
      acc = wmma_f16x32(a, bb, acc);
#pragma unroll
      for (int r = 0; r < 8; ++r) {
        int n = ntile * 16 + r + 8 * hi;
        int m = mtile * 16 + lo;
        float v = acc[r];
        if (n < CGA_N && m < CGA_N)
          v += sBias[bias_idxs[n * CGA_N + m]];
        sAttnH[n][m] = (_Float16)v;
      }
    }
    __syncthreads();

    // ---- P4: row softmax (f32 math on f16 logits), zero pads ----
    if (tid < CGA_N) {
      _Float16* row = &sAttnH[tid][0];
      float mx = -3.0e38f;
      for (int m = 0; m < CGA_N; ++m) mx = fmaxf(mx, (float)row[m]);
      float sum = 0.f;
      for (int m = 0; m < CGA_N; ++m) {
        float e = __expf((float)row[m] - mx);
        sum += e;
        row[m] = (_Float16)e;
      }
      float inv = 1.f / sum;
      for (int m = 0; m < CGA_N; ++m) row[m] = (_Float16)((float)row[m] * inv);
      for (int m = CGA_N; m < CGA_NK; ++m) row[m] = (_Float16)0.f;
    } else if (tid < CGA_NP) {
      _Float16* row = &sAttnH[tid][0];
      for (int m = 0; m < CGA_NK; ++m) row[m] = (_Float16)0.f;
    }
    __syncthreads();

    // ---- P5: out[d,n] = sum_m v[d,m]*attn[n,m]  4x13 tiles, K=224 ----
    float* outg = out_stage + ((size_t)b * CGA_DIM + h * CGA_VD) * CGA_N;
    for (int t = wave; t < 4 * 13; t += 8) {
      int mt = t / 13, ntile = t % 13;
      v8f acc = {};
#pragma unroll
      for (int k0 = 0; k0 < CGA_NK; k0 += 32) {
        v16h a  = load_a16(&sV[mt * 16 + lo][0], k0, hi);          // A[d][m]=v
        v16h bb = load_bt16(&sAttnH[ntile * 16 + lo][0], k0, hi);  // B[m][n]=attn^T
        acc = wmma_f16x32(a, bb, acc);
      }
#pragma unroll
      for (int r = 0; r < 8; ++r) {
        int d = mt * 16 + r + 8 * hi;
        int n = ntile * 16 + lo;
        sFeatT[n][d] = (_Float16)acc[r];                  // raw output cascades
        if (n < CGA_N) outg[d * CGA_N + n] = fmaxf(acc[r], 0.f);
      }
    }
    __syncthreads();
  }
}

// ---------------------------------------------------------------------------
// Kernel 2: proj GEMM + BN, in-place on d_out (input buffered to LDS first).
// ---------------------------------------------------------------------------
__global__ __launch_bounds__(256)
void cga_proj_kernel(const float* __restrict__ proj_w,
                     const float* __restrict__ proj_g,
                     const float* __restrict__ proj_b,
                     const float* __restrict__ proj_rm,
                     const float* __restrict__ proj_rv,
                     float* __restrict__ io) {
  __shared__ __attribute__((aligned(32))) _Float16 sWp[CGA_DIM][CGA_DIM]; // 128 KB
  __shared__ __attribute__((aligned(32))) _Float16 sInT[CGA_NP][CGA_DIM]; // 104 KB
  __shared__ float sS[CGA_DIM], sSh[CGA_DIM];

  const int b    = blockIdx.x;
  const int tid  = threadIdx.x;
  const int lane = tid & 31;
  const int wave = tid >> 5;
  const int lo   = lane & 15;
  const int hi   = lane >> 4;

  float* iob = io + (size_t)b * CGA_DIM * CGA_N;
  for (int i = tid; i < CGA_DIM * CGA_DIM; i += 256)
    sWp[i >> 8][i & 255] = (_Float16)proj_w[i];
  for (int i = tid; i < CGA_DIM * CGA_N; i += 256) {
    int c = i / CGA_N, n = i % CGA_N;             // coalesced global read
    sInT[n][c] = (_Float16)iob[i];
  }
  for (int i = tid; i < (CGA_NP - CGA_N) * CGA_DIM; i += 256)
    sInT[CGA_N + i / CGA_DIM][i % CGA_DIM] = (_Float16)0.f;
  for (int i = tid; i < CGA_DIM; i += 256) {
    float s = proj_g[i] * rsqrtf(proj_rv[i] + 1e-5f);
    sS[i]  = s;
    sSh[i] = proj_b[i] - proj_rm[i] * s;
  }
  __syncthreads();

  // M=256 (16 tiles), N=208 (13 tiles), K=256 (8 wmma steps)
  for (int t = wave; t < 16 * 13; t += 8) {
    int mt = t / 13, nt = t % 13;
    v8f acc = {};
#pragma unroll
    for (int k0 = 0; k0 < CGA_DIM; k0 += 32) {
      v16h a  = load_a16(&sWp[mt * 16 + lo][0], k0, hi);   // A[o][c]
      v16h bb = load_bt16(&sInT[nt * 16 + lo][0], k0, hi); // B[c][n]
      acc = wmma_f16x32(a, bb, acc);
    }
#pragma unroll
    for (int r = 0; r < 8; ++r) {
      int o = mt * 16 + r + 8 * hi;
      int n = nt * 16 + lo;
      if (n < CGA_N) iob[o * CGA_N + n] = acc[r] * sS[o] + sSh[o];
    }
  }
}

extern "C" void kernel_launch(void* const* d_in, const int* in_sizes, int n_in,
                              void* d_out, int out_size, void* d_ws, size_t ws_size,
                              hipStream_t stream) {
  const float* x        = (const float*)d_in[0];
  const float* qkv_w    = (const float*)d_in[1];
  const float* qkv_g    = (const float*)d_in[2];
  const float* qkv_bb   = (const float*)d_in[3];
  const float* qkv_rm   = (const float*)d_in[4];
  const float* qkv_rv   = (const float*)d_in[5];
  const float* dw_w     = (const float*)d_in[6];
  const float* dw_g     = (const float*)d_in[7];
  const float* dw_bb    = (const float*)d_in[8];
  const float* dw_rm    = (const float*)d_in[9];
  const float* dw_rv    = (const float*)d_in[10];
  const float* proj_w   = (const float*)d_in[11];
  const float* proj_g   = (const float*)d_in[12];
  const float* proj_bb  = (const float*)d_in[13];
  const float* proj_rm  = (const float*)d_in[14];
  const float* proj_rv  = (const float*)d_in[15];
  const float* attn_bia = (const float*)d_in[16];
  const int*   bias_idx = (const int*)d_in[17];
  float* out = (float*)d_out;

  cga_attn_kernel<<<CGA_B, 256, 0, stream>>>(
      x, qkv_w, qkv_g, qkv_bb, qkv_rm, qkv_rv,
      dw_w, dw_g, dw_bb, dw_rm, dw_rv, attn_bia, bias_idx, out);
  cga_proj_kernel<<<CGA_B, 256, 0, stream>>>(
      proj_w, proj_g, proj_bb, proj_rm, proj_rv, out);
}
